// EfficientMHA_46883863003136
// MI455X (gfx1250) — matrix operand order repaired and checked by reading, not compiled, not run on hardware
//
#include <hip/hip_runtime.h>

typedef __attribute__((ext_vector_type(16))) __bf16 v16bf;
typedef __attribute__((ext_vector_type(8)))  __bf16 v8bf;
typedef __attribute__((ext_vector_type(8)))  float  v8f;

#define LOG2E 1.4426950408889634f

static __device__ __forceinline__ v8f wmma_bf16(const v16bf& a, const v16bf& b, const v8f& c) {
  return __builtin_amdgcn_wmma_f32_16x16x32_bf16(false, a, false, b, (short)0, c, false, false);
}

// ---- DPP 16-lane (row) reductions: pure VALU, no LDS traffic -------------
template <int CTRL>
static __device__ __forceinline__ float dpp_mov(float x) {
  return __builtin_bit_cast(float, __builtin_amdgcn_update_dpp(
      0, __builtin_bit_cast(int, x), CTRL, 0xF, 0xF, true));
}
static __device__ __forceinline__ float row_max16(float x) {
  x = fmaxf(x, dpp_mov<0xB1>(x));    // quad_perm [1,0,3,2]  (xor 1)
  x = fmaxf(x, dpp_mov<0x4E>(x));    // quad_perm [2,3,0,1]  (xor 2)
  x = fmaxf(x, dpp_mov<0x141>(x));   // row_half_mirror      (8-group swap)
  x = fmaxf(x, dpp_mov<0x140>(x));   // row_mirror           (16-group swap)
  return x;
}
static __device__ __forceinline__ float row_sum16(float x) {
  x += dpp_mov<0xB1>(x);
  x += dpp_mov<0x4E>(x);
  x += dpp_mov<0x141>(x);
  x += dpp_mov<0x140>(x);
  return x;
}

// ---- CDNA5 async global->LDS copy (ASYNCcnt path) ------------------------
static __device__ __forceinline__ void async_load_b128(uint32_t lds_off, const void* gaddr) {
  asm volatile("global_load_async_to_lds_b128 %0, %1, off"
               :: "v"(lds_off), "v"(gaddr) : "memory");
}
static __device__ __forceinline__ void wait_asynccnt0() {
  asm volatile("s_wait_asynccnt 0x0" ::: "memory");
}
static __device__ __forceinline__ uint32_t lds_addr32(const void* p) {
  // generic LDS pointer: high 32 bits = shared aperture, low 32 bits = LDS offset
  return (uint32_t)(uintptr_t)p;
}

// ---------------------------------------------------------------------------
// GEMM: Y[m,n] = scale * sum_k X[m,k] * W[n,k]   (nn.Linear: x @ W^T)
// M=8192 (B*S), N=K=2048.
// XMODE 0: X fp32 row-major [M,K].  XMODE 1: X bf16 head layout [B,H,S,D].
// OMODE 0: Y bf16 head layout [B,H,S,D] (with scale).  OMODE 1: Y fp32 [M,N].
// ---------------------------------------------------------------------------
template <int XMODE, int OMODE>
__global__ __launch_bounds__(256)
void gemm_xwT(const void* __restrict__ Xv, const float* __restrict__ W,
              void* __restrict__ Yv, float scale)
{
  constexpr int K = 2048, N = 2048;
  constexpr int BM = 128, BN = 128, BK = 32;
  constexpr int LDA = BK + 8;   // 40 halves
  constexpr int LDB = BN + 8;   // 136 halves

  __shared__ __bf16 Alds[BM * LDA];
  __shared__ __bf16 Blds[BK * LDB];

  const int tid  = threadIdx.x;
  const int lane = tid & 31;
  const int w    = tid >> 5;
  const int wm   = w >> 2;
  const int wn   = w & 3;
  const int m0   = blockIdx.y * BM;
  const int n0   = blockIdx.x * BN;
  const int half = lane >> 4;
  const int l16  = lane & 15;

  const int la_row = tid >> 1;
  const int la_col = (tid & 1) * 16;

  v8f acc[4][2] = {};

  for (int k0 = 0; k0 < K; k0 += BK) {
    __syncthreads();
    if (XMODE == 0) {
      const float* X = (const float*)Xv;
      const float4* src = (const float4*)&X[(size_t)(m0 + la_row) * K + k0 + la_col];
      if (k0 + BK < K) __builtin_prefetch((const void*)((const char*)src + BK * 4), 0, 0);
      __bf16* dst = &Alds[la_row * LDA + la_col];
#pragma unroll
      for (int u = 0; u < 4; ++u) {
        float4 f = src[u];
        dst[u * 4 + 0] = (__bf16)f.x; dst[u * 4 + 1] = (__bf16)f.y;
        dst[u * 4 + 2] = (__bf16)f.z; dst[u * 4 + 3] = (__bf16)f.w;
      }
    } else {
      const __bf16* X = (const __bf16*)Xv;
      const int mg = m0 + la_row, kg = k0 + la_col;
      const int b = mg >> 11, s = mg & 2047, h = kg >> 7, d = kg & 127;
      const v8bf* src = (const v8bf*)&X[(((size_t)(b * 16 + h) * 2048) + s) * 128 + d];
      v8bf* dst = (v8bf*)&Alds[la_row * LDA + la_col];
      dst[0] = src[0]; dst[1] = src[1];
    }
    {
      const float4* src = (const float4*)&W[(size_t)(n0 + la_row) * K + k0 + la_col];
      if (k0 + BK < K) __builtin_prefetch((const void*)((const char*)src + BK * 4), 0, 0);
#pragma unroll
      for (int u = 0; u < 4; ++u) {
        float4 f = src[u];
        Blds[(la_col + u * 4 + 0) * LDB + la_row] = (__bf16)f.x;
        Blds[(la_col + u * 4 + 1) * LDB + la_row] = (__bf16)f.y;
        Blds[(la_col + u * 4 + 2) * LDB + la_row] = (__bf16)f.z;
        Blds[(la_col + u * 4 + 3) * LDB + la_row] = (__bf16)f.w;
      }
    }
    __syncthreads();

    v16bf afrag[4], bfrag[2];
#pragma unroll
    for (int i = 0; i < 4; ++i) {
      const __bf16* p = &Alds[(wm * 64 + i * 16 + l16) * LDA + half * 8];
      v8bf* a = (v8bf*)&afrag[i];
      a[0] = *(const v8bf*)p;
      a[1] = *(const v8bf*)(p + 16);
    }
#pragma unroll
    for (int j = 0; j < 2; ++j) {
      const __bf16* p = &Blds[half * 8 * LDB + wn * 32 + j * 16 + l16];
#pragma unroll
      for (int s = 0; s < 16; ++s) bfrag[j][s] = p[((s & 7) + (s >> 3) * 16) * LDB];
    }
#pragma unroll
    for (int i = 0; i < 4; ++i)
#pragma unroll
      for (int j = 0; j < 2; ++j)
        acc[i][j] = wmma_bf16(afrag[i], bfrag[j], acc[i][j]);
  }

#pragma unroll
  for (int i = 0; i < 4; ++i)
#pragma unroll
    for (int j = 0; j < 2; ++j)
#pragma unroll
      for (int r = 0; r < 8; ++r) {
        const int m = m0 + wm * 64 + i * 16 + r + half * 8;
        const int n = n0 + wn * 32 + j * 16 + l16;
        const float v = acc[i][j][r] * scale;
        if (OMODE == 0) {
          const int b = m >> 11, s = m & 2047, h = n >> 7, d = n & 127;
          ((__bf16*)Yv)[(((size_t)(b * 16 + h) * 2048) + s) * 128 + d] = (__bf16)v;
        } else {
          ((float*)Yv)[(size_t)m * N + n] = v;
        }
      }
}

// ---------------------------------------------------------------------------
// Flash attention: block = (b*h, 128 query rows), 8 waves; each wave owns 16
// query rows with a private online softmax over 2048 keys in 64-key tiles.
// Q/K/V/O bf16 [B,H,S,D], Q pre-scaled by 1/sqrt(D).
// ---------------------------------------------------------------------------
__global__ __launch_bounds__(256)
void flash_attn(const __bf16* __restrict__ Q, const __bf16* __restrict__ Kh,
                const __bf16* __restrict__ V, __bf16* __restrict__ O)
{
  constexpr int S = 2048, Dh = 128, KB = 64, QB = 128;
  constexpr int LKT = KB + 8;   // Kt[d][n], 72 halves (144B rows, 16B aligned)
  constexpr int LV  = Dh + 8;   // Vl[n][d], 136 halves
  constexpr int LP  = KB + 8;   // per-wave P slab, 72 halves

  __shared__ __bf16 Kt[Dh * LKT];       // transposed K tile (d-major)
  __shared__ __bf16 Vl[KB * LV];        // V tile row-major (async-filled)
  __shared__ __bf16 Pl[8 * 16 * LP];    // per-wave P bounce buffers

  const int tid  = threadIdx.x;
  const int lane = tid & 31;
  const int w    = tid >> 5;
  const int bh   = blockIdx.x >> 4;
  const int qb   = blockIdx.x & 15;
  const int half = lane >> 4;
  const int l16  = lane & 15;
  const size_t base = (size_t)bh * S * Dh;

  // resident Q A-fragments (16 rows x 128d = 4 k-steps)
  v16bf aq[4];
  {
    const size_t qrow = base + (size_t)(qb * QB + w * 16 + l16) * Dh;
#pragma unroll
    for (int dd = 0; dd < 4; ++dd) {
      const __bf16* p = &Q[qrow + dd * 32 + half * 8];
      v8bf* a = (v8bf*)&aq[dd];
      a[0] = *(const v8bf*)p;
      a[1] = *(const v8bf*)(p + 16);
    }
  }

  v8f o[8] = {};
  float mi[8], li[8];
#pragma unroll
  for (int r = 0; r < 8; ++r) { mi[r] = -1e30f; li[r] = 0.0f; }

  const int ln = tid >> 2;        // 0..63: K/V tile row
  const int ld = (tid & 3) * 32;  // d segment base (32 halves)

  for (int kv = 0; kv < S; kv += KB) {
    __syncthreads();
    // ---- V tile: async global->LDS, row-major (exactly the B-frag layout) --
    {
      const __bf16* g = &V[base + (size_t)(kv + ln) * Dh + ld];
      const uint32_t loff = lds_addr32(&Vl[ln * LV + ld]);
#pragma unroll
      for (int u = 0; u < 4; ++u)
        async_load_b128(loff + u * 16, (const void*)(g + u * 8));
    }
    // ---- K tile: stage transposed Kt[d][n] through registers ----
    {
      const v8bf* src = (const v8bf*)&Kh[base + (size_t)(kv + ln) * Dh + ld];
      if (kv + KB < S) __builtin_prefetch((const void*)((const char*)src + KB * Dh * 2), 0, 0);
#pragma unroll
      for (int u = 0; u < 4; ++u) {
        v8bf rr = src[u];
#pragma unroll
        for (int e = 0; e < 8; ++e) Kt[(ld + u * 8 + e) * LKT + ln] = rr[e];
      }
    }
    wait_asynccnt0();
    __syncthreads();

    // ---- scores = Q (16xD) @ K^T (D x 64): 4 accum tiles, 16 WMMA ----
    v8f sc[4] = {};
#pragma unroll
    for (int dd = 0; dd < 4; ++dd)
#pragma unroll
      for (int j = 0; j < 4; ++j) {
        v16bf bk;
        const __bf16* p = &Kt[(dd * 32 + half * 8) * LKT + j * 16 + l16];
#pragma unroll
        for (int s = 0; s < 16; ++s) bk[s] = p[((s & 7) + (s >> 3) * 16) * LKT];
        sc[j] = wmma_bf16(aq[dd], bk, sc[j]);
      }

    // ---- online softmax per C-layout row (r + 8*half), DPP reductions ----
#pragma unroll
    for (int r = 0; r < 8; ++r) {
      float s0 = sc[0][r], s1 = sc[1][r], s2 = sc[2][r], s3 = sc[3][r];
      const float mx = row_max16(fmaxf(fmaxf(s0, s1), fmaxf(s2, s3)));
      const float mnew  = fmaxf(mi[r], mx);
      const float alpha = exp2f((mi[r] - mnew) * LOG2E);
      const float p0 = exp2f((s0 - mnew) * LOG2E);
      const float p1 = exp2f((s1 - mnew) * LOG2E);
      const float p2 = exp2f((s2 - mnew) * LOG2E);
      const float p3 = exp2f((s3 - mnew) * LOG2E);
      const float rs = row_sum16((p0 + p1) + (p2 + p3));
      li[r] = li[r] * alpha + rs;
      mi[r] = mnew;
#pragma unroll
      for (int dt = 0; dt < 8; ++dt) o[dt][r] *= alpha;
      const int prow = w * 16 + r + half * 8;
      Pl[prow * LP + 0 * 16 + l16] = (__bf16)p0;
      Pl[prow * LP + 1 * 16 + l16] = (__bf16)p1;
      Pl[prow * LP + 2 * 16 + l16] = (__bf16)p2;
      Pl[prow * LP + 3 * 16 + l16] = (__bf16)p3;
    }

    // ---- acc += P (16x64) @ V (64xD): 2 k-steps x 8 d-tiles = 16 WMMA ----
    v16bf ap[2];
#pragma unroll
    for (int f = 0; f < 2; ++f) {
      const __bf16* p = &Pl[(w * 16 + l16) * LP + f * 32 + half * 8];
      v8bf* a = (v8bf*)&ap[f];
      a[0] = *(const v8bf*)p;
      a[1] = *(const v8bf*)(p + 16);
    }
#pragma unroll
    for (int dt = 0; dt < 8; ++dt) {
#pragma unroll
      for (int f = 0; f < 2; ++f) {
        v16bf bv;
        const __bf16* p = &Vl[(f * 32 + half * 8) * LV + dt * 16 + l16];
#pragma unroll
        for (int s = 0; s < 16; ++s) bv[s] = p[((s & 7) + (s >> 3) * 16) * LV];
        o[dt] = wmma_bf16(ap[f], bv, o[dt]);
      }
    }
  }

  // ---- normalize + store O bf16 [B,H,S,D] ----
#pragma unroll
  for (int r = 0; r < 8; ++r) {
    const float inv = 1.0f / li[r];
    const int row = qb * QB + w * 16 + r + half * 8;
#pragma unroll
    for (int dt = 0; dt < 8; ++dt)
      O[base + (size_t)row * Dh + dt * 16 + l16] = (__bf16)(o[dt][r] * inv);
  }
}

// ---------------------------------------------------------------------------
extern "C" void kernel_launch(void* const* d_in, const int* in_sizes, int n_in,
                              void* d_out, int out_size, void* d_ws, size_t ws_size,
                              hipStream_t stream) {
  (void)in_sizes; (void)n_in; (void)out_size; (void)ws_size;
  const float* q  = (const float*)d_in[0];
  const float* k  = (const float*)d_in[1];
  const float* v  = (const float*)d_in[2];
  const float* Wq = (const float*)d_in[3];
  const float* Wk = (const float*)d_in[4];
  const float* Wv = (const float*)d_in[5];
  const float* Wo = (const float*)d_in[6];

  const size_t seg = (size_t)4 * 16 * 2048 * 128 * sizeof(__bf16);
  __bf16* Qbf = (__bf16*)d_ws;
  __bf16* Kbf = (__bf16*)((char*)d_ws + seg);
  __bf16* Vbf = (__bf16*)((char*)d_ws + 2 * seg);
  __bf16* Obf = (__bf16*)((char*)d_ws + 3 * seg);

  dim3 block(256);
  dim3 ggrid(2048 / 128, 8192 / 128);

  const float qscale = 0.08838834764831845f;  // 1/sqrt(128)
  gemm_xwT<0, 0><<<ggrid, block, 0, stream>>>((const void*)q, Wq, (void*)Qbf, qscale);
  gemm_xwT<0, 0><<<ggrid, block, 0, stream>>>((const void*)k, Wk, (void*)Kbf, 1.0f);
  gemm_xwT<0, 0><<<ggrid, block, 0, stream>>>((const void*)v, Wv, (void*)Vbf, 1.0f);

  flash_attn<<<dim3(64 * 16), block, 0, stream>>>(Qbf, Kbf, Vbf, Obf);

  gemm_xwT<1, 1><<<ggrid, block, 0, stream>>>((const void*)Obf, Wo, d_out, 1.0f);
}
